// GPT2Attention_15728170238309
// MI455X (gfx1250) — matrix operand order repaired and checked by reading, not compiled, not run on hardware
//
#include <hip/hip_runtime.h>
#include <hip/hip_bf16.h>

// ---------------------------------------------------------------------------
// GPT-2 attention forward on gfx1250 (MI455X), bf16 WMMA + async LDS copies.
//   T = 2048, D = 2048, H = 16, d_head = 128
//   k0a: x -> bf16          k0b/k0c: W_attn / W_proj -> bf16, TRANSPOSED [N][K]
//   k1 : c = x @ W_attn + b_attn  (128x128 tile GEMM, double-buffered async)
//   k2 : flash attention per (head, 64-query-row tile)
//   k3 : a = merged @ W_proj + b_proj
// ---------------------------------------------------------------------------

typedef __attribute__((ext_vector_type(16))) __bf16 v16bf;
typedef __attribute__((ext_vector_type(8)))  float  v8f;

#define Tdim   2048
#define Ddim   2048
#define N3dim  6144
#define Hheads 16
#define DH     128

// ------------------------- async global->LDS copy --------------------------
// ISA 15.18.3 op 98: GLOBAL_LOAD_ASYNC_TO_LDS_B128 (tracked with ASYNCcnt).
// vdst = LDS byte address (VGPR), vaddr = 64-bit global address (VGPR pair).
// Generic LDS pointers carry the LDS offset in their low 32 bits (ISA 10.2).
static __device__ __forceinline__ void async_copy_b128(void* lds_dst, const void* gsrc) {
#if defined(__gfx1250__)
  const unsigned ldsoff = (unsigned)(unsigned long long)(size_t)lds_dst;
  asm volatile("global_load_async_to_lds_b128 %0, %1, off"
               :: "v"(ldsoff), "v"((unsigned long long)(size_t)gsrc)
               : "memory");
#else
  *(uint4*)lds_dst = *(const uint4*)gsrc;
#endif
}

static __device__ __forceinline__ void async_wait_all() {
#if defined(__gfx1250__)
  asm volatile("s_wait_asynccnt 0" ::: "memory");
#endif
}

// ------------------------------- helpers -----------------------------------
static __device__ __forceinline__ __bf16 f2bf(float f) {
  union { float f; unsigned u; } c; c.f = f;
  unsigned u = c.u;
  unsigned r = u + 0x7FFFu + ((u >> 16) & 1u);   // round-to-nearest-even
  union { unsigned short s; __bf16 b; } o;
  o.s = (unsigned short)(r >> 16);
  return o.b;
}

// A-fragment 16x32 bf16 (ISA 7.12.2): lane L holds row M=L&15.
// Two contiguous 16B chunks: K=[kb,kb+8) and K=[16+kb,16+kb+8), kb=8*(lane>=16).
// All LDS strides used are multiples of 16B -> ds_load_b128 x2.
static __device__ __forceinline__ v16bf load_a_frag(const __bf16* p0, int lda) {
  const int lane = threadIdx.x & 31;
  const __bf16* p = p0 + (size_t)(lane & 15) * lda + ((lane >> 4) << 3);
  union { uint4 u[2]; v16bf v; } x;
  x.u[0] = *(const uint4*)(p);
  x.u[1] = *(const uint4*)(p + 16);
  return x.v;
}

// B-fragment 32x16 bf16 from an LDS tile stored TRANSPOSED as [N][K]:
// Same K order as A: chunks K=[kb,kb+8) and K=[16+kb,16+kb+8), kb=8*(lane>=16).
static __device__ __forceinline__ v16bf load_b_frag(const __bf16* pT0, int ldk) {
  const int lane = threadIdx.x & 31;
  const __bf16* p = pT0 + (size_t)(lane & 15) * ldk + ((lane >> 4) << 3);
  union { uint4 u[2]; v16bf v; } x;
  x.u[0] = *(const uint4*)(p);
  x.u[1] = *(const uint4*)(p + 16);
  return x.v;
}

static __device__ __forceinline__ v8f wmma_bf16(v16bf a, v16bf b, v8f c) {
  return __builtin_amdgcn_wmma_f32_16x16x32_bf16(false, a, false, b,
                                                 (short)0, c, false, false);
}

// ---------------------------------------------------------------------------
// k0a: elementwise fp32 -> bf16  (x)
// ---------------------------------------------------------------------------
__global__ __launch_bounds__(256) void convert_kernel(const float* __restrict__ src,
                                                      __bf16* __restrict__ dst) {
  const size_t i = ((size_t)blockIdx.x * 256 + threadIdx.x) * 8;
#pragma unroll
  for (int j = 0; j < 8; ++j) dst[i + j] = f2bf(src[i + j]);
}

// ---------------------------------------------------------------------------
// k0b/k0c: W[K=2048][Ncols] fp32  ->  Wt[Ncols][2048] bf16  (LDS-tiled)
// ---------------------------------------------------------------------------
__global__ __launch_bounds__(256) void transpose_kernel(const float* __restrict__ W,
                                                        __bf16* __restrict__ Wt,
                                                        int Ncols) {
  __shared__ float tile[32][33];
  const int t = threadIdx.x;
  const int tx = t & 31, ty = t >> 5;            // 32 x 8
  const int n0 = blockIdx.x * 32, k0 = blockIdx.y * 32;
#pragma unroll
  for (int i = 0; i < 4; ++i)
    tile[ty + 8 * i][tx] = W[(size_t)(k0 + ty + 8 * i) * Ncols + n0 + tx];
  __syncthreads();
#pragma unroll
  for (int i = 0; i < 4; ++i)
    Wt[(size_t)(n0 + ty + 8 * i) * Ddim + k0 + tx] = f2bf(tile[tx][ty + 8 * i]);
}

// ---------------------------------------------------------------------------
// GEMM: C[M,N] = A[M,2048]bf16 @ Bt[N,2048]bf16^T + bias
// Block tile 128x128, BK=32, 256 threads = 8 waves, 32x64 per wave (8 WMMA/step)
// Double-buffered LDS: async-prefetch k+32 while computing k.
// WRITE_QKV=1: also store bf16 c[T,3D] + spill K/V fp32 into new_past.
// ---------------------------------------------------------------------------
template <int WRITE_QKV>
__global__ __launch_bounds__(256) void gemm_kernel(const __bf16* __restrict__ A,
                                                   const __bf16* __restrict__ Bt,
                                                   const float* __restrict__ bias,
                                                   float* __restrict__ outF,
                                                   __bf16* __restrict__ outBF) {
  __shared__ struct {
    __bf16 As[2][128][40];   // [buf][M][K] padded (80B row stride, 16B aligned)
    __bf16 Bs[2][128][40];   // [buf][N][K] padded
  } sm;

  const int m0 = blockIdx.y * 128;
  const int n0 = blockIdx.x * 128;
  const int t = threadIdx.x;
  const int lane = t & 31;
  const int w = t >> 5;
  const int mrow = (w & 3) * 32;   // wave's 32 rows
  const int ncol = (w >> 2) * 64;  // wave's 64 cols

  v8f acc[2][4];
#pragma unroll
  for (int s = 0; s < 2; ++s)
#pragma unroll
    for (int j = 0; j < 4; ++j)
#pragma unroll
      for (int r = 0; r < 8; ++r) acc[s][j][r] = 0.0f;

  const int srow = t >> 1;           // staging row 0..127
  const int skb  = (t & 1) * 16;     // staging k-offset (elements)

  auto stage = [&](int buf, int k0) {
    const __bf16* ga = A  + (size_t)(m0 + srow) * Ddim + k0 + skb;
    const __bf16* gb = Bt + (size_t)(n0 + srow) * Ddim + k0 + skb;
    async_copy_b128(&sm.As[buf][srow][skb],     ga);
    async_copy_b128(&sm.As[buf][srow][skb + 8], ga + 8);
    async_copy_b128(&sm.Bs[buf][srow][skb],     gb);
    async_copy_b128(&sm.Bs[buf][srow][skb + 8], gb + 8);
  };

  stage(0, 0);
  async_wait_all();
  __syncthreads();

  for (int k0 = 0; k0 < Ddim; k0 += 32) {
    const int cur = (k0 >> 5) & 1;
    if (k0 + 32 < Ddim) stage(cur ^ 1, k0 + 32);   // async prefetch next block

    v16bf a0 = load_a_frag(&sm.As[cur][mrow][0], 40);
    v16bf a1 = load_a_frag(&sm.As[cur][mrow + 16][0], 40);
#pragma unroll
    for (int j = 0; j < 4; ++j) {
      v16bf b = load_b_frag(&sm.Bs[cur][ncol + j * 16][0], 40);
      acc[0][j] = wmma_bf16(a0, b, acc[0][j]);
      acc[1][j] = wmma_bf16(a1, b, acc[1][j]);
    }
    async_wait_all();   // next buffer landed
    __syncthreads();    // all waves done with `cur` before it is re-staged
  }

  const int lanehalf = (lane >> 4) << 3;
#pragma unroll
  for (int s = 0; s < 2; ++s) {
    const int rbase = m0 + mrow + s * 16 + lanehalf;
#pragma unroll
    for (int j = 0; j < 4; ++j) {
      const int col = n0 + ncol + j * 16 + (lane & 15);
      const float bv = bias[col];
#pragma unroll
      for (int r = 0; r < 8; ++r) {
        const int row = rbase + r;
        const float v = acc[s][j][r] + bv;
        if (WRITE_QKV) {
          outBF[(size_t)row * N3dim + col] = f2bf(v);
          if (col >= Ddim) {                           // spill K/V fp32
            const int cc = col - Ddim;
            const int which = cc >> 11;                // 0 = k, 1 = v
            const int hh = (cc & (Ddim - 1)) >> 7;
            const int jj = cc & (DH - 1);
            const size_t off = (size_t)Tdim * Ddim +
                               (size_t)which * Hheads * Tdim * DH +
                               (size_t)hh * Tdim * DH + (size_t)row * DH + jj;
            outF[off] = v;
          }
        } else {
          outF[(size_t)row * Ddim + col] = v;
        }
      }
    }
  }
}

// ---------------------------------------------------------------------------
// k2: flash attention, one (head, 64-query-row) tile per workgroup.
// ---------------------------------------------------------------------------
__global__ __launch_bounds__(256) void attn_kernel(const __bf16* __restrict__ c_bf,
                                                   __bf16* __restrict__ a_bf) {
  __shared__ struct {
    __bf16 Qs[64][136];   // [M][d]       (272B stride, 16B aligned)
    __bf16 Ks[64][136];   // [Nk][d]  == B^T layout for Q*K^T
    __bf16 Vt[128][72];   // [d][Nk]  == B^T layout for P*V (144B stride)
    float  Sf[64][68];    // fp32 scores
    __bf16 Ps[64][72];    // probabilities (A operand)
    float  rowMax[64], rowSum[64], rowScale[64];
  } sm;

  const int h  = blockIdx.y;
  const int m0 = blockIdx.x * 64;
  const int t = threadIdx.x;
  const int lane = t & 31;
  const int w = t >> 5;
  const int mt = w & 3;
  const int ng = w >> 2;

  const int srow = t >> 2;          // staging row 0..63
  const int sseg = (t & 3) * 32;    // staging col segment (elements)

  {  // stage Q once (async, 4x b128 per thread)
    const __bf16* src = c_bf + (size_t)(m0 + srow) * N3dim + h * DH + sseg;
#pragma unroll
    for (int i = 0; i < 4; ++i)
      async_copy_b128(&sm.Qs[srow][sseg + 8 * i], src + 8 * i);
  }
  if (t < 64) { sm.rowMax[t] = -3.0e38f; sm.rowSum[t] = 0.0f; }

  v8f oacc[4];
#pragma unroll
  for (int j = 0; j < 4; ++j)
#pragma unroll
    for (int r = 0; r < 8; ++r) oacc[j][r] = 0.0f;

  const float rscale = 0.08838834764831845f;  // 1/sqrt(128)

  for (int j0 = 0; j0 <= m0; j0 += 64) {
    __syncthreads();  // previous iteration done with Ks/Vt/Ps
    {  // stage K (async) and V (manual transpose)
      const __bf16* srck = c_bf + (size_t)(j0 + srow) * N3dim + Ddim + h * DH + sseg;
      const __bf16* srcv = c_bf + (size_t)(j0 + srow) * N3dim + 2 * Ddim + h * DH + sseg;
#pragma unroll
      for (int i = 0; i < 4; ++i)
        async_copy_b128(&sm.Ks[srow][sseg + 8 * i], srck + 8 * i);
#pragma unroll
      for (int i = 0; i < 32; ++i) sm.Vt[sseg + i][srow] = srcv[i];
    }
    async_wait_all();
    __syncthreads();

    // S = Q * K^T   (64x64, K-loop d=128)
    v8f sacc[2];
#pragma unroll
    for (int j = 0; j < 2; ++j)
#pragma unroll
      for (int r = 0; r < 8; ++r) sacc[j][r] = 0.0f;
#pragma unroll
    for (int kk = 0; kk < 4; ++kk) {
      v16bf a = load_a_frag(&sm.Qs[mt * 16][kk * 32], 136);
#pragma unroll
      for (int j = 0; j < 2; ++j) {
        v16bf b = load_b_frag(&sm.Ks[ng * 32 + j * 16][kk * 32], 136);
        sacc[j] = wmma_bf16(a, b, sacc[j]);
      }
    }

    // scale + causal mask -> Sf
    {
      const int lr = mt * 16 + ((lane >> 4) << 3);
      const int lc = ng * 32 + (lane & 15);
#pragma unroll
      for (int j = 0; j < 2; ++j)
#pragma unroll
        for (int r = 0; r < 8; ++r) {
          const int row = lr + r;
          const int col = lc + j * 16;
          float v = sacc[j][r] * rscale;
          if (j0 + col > m0 + row) v = -1.0e30f;
          sm.Sf[row][col] = v;
        }
    }
    __syncthreads();

    // online softmax row update (64 rows, one per thread)
    if (t < 64) {
      float mold = sm.rowMax[t];
      float mnew = mold;
      for (int c = 0; c < 64; ++c) mnew = fmaxf(mnew, sm.Sf[t][c]);
      const float scale = __expf(mold - mnew);
      float s = sm.rowSum[t] * scale;
      for (int c = 0; c < 64; ++c) {
        const float p = __expf(sm.Sf[t][c] - mnew);
        s += p;
        sm.Ps[t][c] = f2bf(p);
      }
      sm.rowMax[t] = mnew; sm.rowSum[t] = s; sm.rowScale[t] = scale;
    }
    __syncthreads();

    // rescale running O, then O += P * V
    {
      const int mr = mt * 16 + ((lane >> 4) << 3);
#pragma unroll
      for (int j = 0; j < 4; ++j)
#pragma unroll
        for (int r = 0; r < 8; ++r) oacc[j][r] *= sm.rowScale[mr + r];
    }
#pragma unroll
    for (int kk = 0; kk < 2; ++kk) {
      v16bf a = load_a_frag(&sm.Ps[mt * 16][kk * 32], 72);
#pragma unroll
      for (int j = 0; j < 4; ++j) {
        v16bf b = load_b_frag(&sm.Vt[(ng * 4 + j) * 16][kk * 32], 72);
        oacc[j] = wmma_bf16(a, b, oacc[j]);
      }
    }
  }

  // normalize + write merged-head bf16 output
  {
    const int mr = mt * 16 + ((lane >> 4) << 3);
    const int cb = ng * 64 + (lane & 15);
#pragma unroll
    for (int j = 0; j < 4; ++j)
#pragma unroll
      for (int r = 0; r < 8; ++r) {
        const int row = m0 + mr + r;
        const float inv = 1.0f / sm.rowSum[mr + r];
        a_bf[(size_t)row * Ddim + h * DH + cb + j * 16] = f2bf(oacc[j][r] * inv);
      }
  }
}

// ---------------------------------------------------------------------------
extern "C" void kernel_launch(void* const* d_in, const int* in_sizes, int n_in,
                              void* d_out, int out_size, void* d_ws, size_t ws_size,
                              hipStream_t stream) {
  (void)in_sizes; (void)n_in; (void)out_size; (void)ws_size;

  const float* x      = (const float*)d_in[0];
  const float* W_attn = (const float*)d_in[1];
  const float* b_attn = (const float*)d_in[2];
  const float* W_proj = (const float*)d_in[3];
  const float* b_proj = (const float*)d_in[4];

  float* out = (float*)d_out;

  // workspace layout (bytes)
  char* ws = (char*)d_ws;
  __bf16* c_bf = (__bf16*)(ws);                                        // T*3D
  __bf16* a_bf = (__bf16*)(ws + (size_t)Tdim * N3dim * 2);             // T*D
  __bf16* xb   = (__bf16*)(ws + (size_t)(Tdim * N3dim + Tdim * Ddim) * 2);
  __bf16* WtA  = (__bf16*)(ws + (size_t)(Tdim * N3dim + 2 * Tdim * Ddim) * 2);
  __bf16* WtP  = (__bf16*)(ws + (size_t)(2 * Tdim * N3dim + 2 * Tdim * Ddim) * 2);

  dim3 blk(256, 1, 1);

  convert_kernel<<<dim3((Tdim * Ddim) / (256 * 8)), blk, 0, stream>>>(x, xb);
  transpose_kernel<<<dim3(N3dim / 32, Ddim / 32), blk, 0, stream>>>(W_attn, WtA, N3dim);
  transpose_kernel<<<dim3(Ddim / 32, Ddim / 32), blk, 0, stream>>>(W_proj, WtP, Ddim);

  gemm_kernel<1><<<dim3(N3dim / 128, Tdim / 128), blk, 0, stream>>>(xb, WtA, b_attn, out, c_bf);
  attn_kernel<<<dim3(Tdim / 64, Hheads), blk, 0, stream>>>(c_bf, a_bf);
  gemm_kernel<0><<<dim3(Ddim / 128, Tdim / 128), blk, 0, stream>>>(a_bf, WtP, b_proj, out, nullptr);
}
